// NATFusionModule_1769526526630
// MI455X (gfx1250) — matrix operand order repaired and checked by reading, not compiled, not run on hardware
//
#include <hip/hip_runtime.h>

// ---------------- problem constants (from reference) ----------------
#define B_  2
#define H_  64
#define W_  64
#define C_  256
#define NH_ 8
#define HD_ 32
#define KW_ 7               // kernel_size
#define M_  (B_ * H_ * W_)  // 8192 rows for all GEMMs

typedef __bf16 bf16;
typedef __attribute__((ext_vector_type(16))) __bf16 v16bf;
typedef __attribute__((ext_vector_type(8)))  __bf16 v8bf;
typedef __attribute__((ext_vector_type(8)))  float  v8f;

// ---------------- f32 -> bf16 elementwise (8 elems / thread) ----------------
__global__ __launch_bounds__(256) void f32_to_bf16_vec(
    const float* __restrict__ in, bf16* __restrict__ out, int n8) {
  int i = blockIdx.x * blockDim.x + threadIdx.x;
  if (i >= n8) return;
  const float4* p = (const float4*)in + (size_t)i * 2;
  float4 a = p[0], b = p[1];
  v8bf o;
  o[0] = (bf16)a.x; o[1] = (bf16)a.y; o[2] = (bf16)a.z; o[3] = (bf16)a.w;
  o[4] = (bf16)b.x; o[5] = (bf16)b.y; o[6] = (bf16)b.z; o[7] = (bf16)b.w;
  *((v8bf*)out + i) = o;
}

// ---------------- 256x256 transpose, f32 -> bf16 : WT[n][k] = W[k][n] -------
__global__ __launch_bounds__(256) void transpose_to_bf16_256(
    const float* __restrict__ Win, bf16* __restrict__ WT) {
  __shared__ float tile[16][17];
  int kb = blockIdx.x * 16, nb = blockIdx.y * 16;
  int tx = threadIdx.x, ty = threadIdx.y;
  tile[ty][tx] = Win[(size_t)(kb + ty) * C_ + (nb + tx)];
  __syncthreads();
  WT[(size_t)(nb + ty) * C_ + (kb + tx)] = (bf16)tile[tx][ty];
}

// ---------------- WMMA GEMM: Out = alpha * (A @ B + bias) -------------------
// A  : M x 256  bf16 row-major
// BT : N x 256  bf16 row-major (B transposed so B-fragments are contiguous)
// One wave computes a 32(M) x 64(N) tile, register double-buffered over K.
template <bool OUT_BF16>
__global__ __launch_bounds__(256) void gemm_wmma_bf16(
    const bf16* __restrict__ A, const bf16* __restrict__ BT,
    const float* __restrict__ bias, void* __restrict__ Out,
    int M, int N, float alpha) {
  constexpr int KD = C_;  // K = 256 for every GEMM in this net
  const int wave = (blockIdx.x * blockDim.x + threadIdx.x) >> 5;
  const int lane = threadIdx.x & 31;
  const int half = lane >> 4;   // 0: lanes 0-15, 1: lanes 16-31
  const int lrow = lane & 15;

  const int ntn = N >> 6;                  // 64-wide N strips
  const int m0 = (wave / ntn) * 32;
  const int n0 = (wave % ntn) * 64;
  if (m0 >= M) return;

  const bf16* Ar0 = A + (size_t)(m0 + lrow) * KD;
  const bf16* Ar1 = A + (size_t)(m0 + 16 + lrow) * KD;
  const bf16* Br[4];
#pragma unroll
  for (int t = 0; t < 4; ++t)
    Br[t] = BT + (size_t)(n0 + t * 16 + lrow) * KD + half * 16;

  // A layout (16x32), contiguous k to match B: lanes 0-15 hold K{0..15},
  // lanes 16-31 hold K{16..31}.
  auto loadA = [&](const bf16* r, int k0) {
    v8bf lo = *(const v8bf*)(r + k0 + half * 16);
    v8bf hi = *(const v8bf*)(r + k0 + half * 16 + 8);
    v16bf f;
#pragma unroll
    for (int e = 0; e < 8; ++e) { f[e] = lo[e]; f[8 + e] = hi[e]; }
    return f;
  };

  v8f acc[2][4] = {};
  v16bf a0 = loadA(Ar0, 0);
  v16bf a1 = loadA(Ar1, 0);
  v16bf bfr[4];
#pragma unroll
  for (int t = 0; t < 4; ++t) bfr[t] = *(const v16bf*)(Br[t]);

#pragma unroll
  for (int k0 = 0; k0 < KD; k0 += 32) {
    v16bf a0n, a1n, bn[4];
    const bool more = (k0 + 32) < KD;
    if (more) {                       // issue next-slice loads first ...
      __builtin_prefetch(Ar0 + k0 + 64, 0, 3);   // near-cache prefetch
      a0n = loadA(Ar0, k0 + 32);
      a1n = loadA(Ar1, k0 + 32);
#pragma unroll
      for (int t = 0; t < 4; ++t) bn[t] = *(const v16bf*)(Br[t] + k0 + 32);
    }
#pragma unroll
    for (int t = 0; t < 4; ++t) {     // ... then compute on current buffers
      acc[0][t] = __builtin_amdgcn_wmma_f32_16x16x32_bf16(
          false, a0, false, bfr[t], (short)0, acc[0][t], false, false);
      acc[1][t] = __builtin_amdgcn_wmma_f32_16x16x32_bf16(
          false, a1, false, bfr[t], (short)0, acc[1][t], false, false);
    }
    if (more) {
      a0 = a0n; a1 = a1n;
#pragma unroll
      for (int t = 0; t < 4; ++t) bfr[t] = bn[t];
    }
  }

  // C/D f32 layout: lane holds rows m0 + mt*16 + r + 8*half, col n0' + lrow.
#pragma unroll
  for (int mt = 0; mt < 2; ++mt)
#pragma unroll
    for (int t = 0; t < 4; ++t) {
      int n = n0 + t * 16 + lrow;
      float bs = bias[n];
#pragma unroll
      for (int r = 0; r < 8; ++r) {
        int m = m0 + mt * 16 + r + half * 8;
        float val = alpha * (acc[mt][t][r] + bs);
        if (OUT_BF16)
          ((bf16*)Out)[(size_t)m * N + n] = (bf16)val;
        else
          ((float*)Out)[(size_t)m * N + n] = val;
      }
    }
}

// ---------------- neighborhood attention, LDS-staged window -----------------
// One block = one (b, head, i) row of 64 queries; the 7 key/value rows the
// whole row needs are staged once in LDS (2 x 28 KB bf16), then each thread
// runs an online-softmax over its own 7x7 window out of LDS.
__global__ __launch_bounds__(64) void nat_attention(
    const bf16* __restrict__ qb, const bf16* __restrict__ kb,
    const bf16* __restrict__ vb, const float* __restrict__ rpb,
    bf16* __restrict__ ob) {
  __shared__ v8bf ksh[KW_ * W_ * HD_ / 8];   // 28 KB
  __shared__ v8bf vsh[KW_ * W_ * HD_ / 8];   // 28 KB

  const int i  = blockIdx.x & (H_ - 1);
  const int nh = (blockIdx.x >> 6) & (NH_ - 1);
  const int b  = blockIdx.x >> 9;
  const int j  = threadIdx.x;

  int rs = i - KW_ / 2; rs = rs < 0 ? 0 : (rs > H_ - KW_ ? H_ - KW_ : rs);

  // cooperative stage of K/V rows rs..rs+6, all 64 cols, this head's 32 dims
  const int CH = KW_ * W_ * HD_ / 8;  // 1792 16-byte chunks per tensor
  for (int c = threadIdx.x; c < CH; c += 64) {
    int p   = c >> 8;          // 256 chunks per key row
    int rem = c & 255;
    int col = rem >> 2;
    int dc  = rem & 3;
    size_t g = ((size_t)((b * H_ + rs + p) * W_ + col)) * (C_ / 8) + nh * 4 + dc;
    ksh[c] = ((const v8bf*)kb)[g];
    vsh[c] = ((const v8bf*)vb)[g];
  }
  __syncthreads();

  const size_t qi = ((size_t)((b * H_ + i) * W_ + j)) * (C_ / 8) + nh * 4;
  float qv[HD_];
#pragma unroll
  for (int c = 0; c < 4; ++c) {
    v8bf qc = ((const v8bf*)qb)[qi + c];
#pragma unroll
    for (int e = 0; e < 8; ++e) qv[c * 8 + e] = (float)qc[e];
  }

  int cs = j - KW_ / 2; cs = cs < 0 ? 0 : (cs > W_ - KW_ ? W_ - KW_ : cs);
  const float* rp = rpb + (size_t)nh * (2 * KW_ - 1) * (2 * KW_ - 1);

  float mmax = -3.0e38f, lsum = 0.0f;
  float ov[HD_];
#pragma unroll
  for (int d = 0; d < HD_; ++d) ov[d] = 0.0f;

#pragma unroll 1
  for (int p = 0; p < KW_; ++p) {
    int ridx = rs + p - i + (KW_ - 1);
#pragma unroll 1
    for (int q = 0; q < KW_; ++q) {
      int cj = cs + q;
      int cidx = cj - j + (KW_ - 1);
      int base = (p * W_ + cj) * (HD_ / 8);

      float s = rp[ridx * (2 * KW_ - 1) + cidx];
#pragma unroll
      for (int c = 0; c < 4; ++c) {
        v8bf kc = ksh[base + c];
#pragma unroll
        for (int e = 0; e < 8; ++e) s += qv[c * 8 + e] * (float)kc[e];
      }

      float mnew = s > mmax ? s : mmax;
      float corr = __expf(mmax - mnew);
      float w    = __expf(s - mnew);
      lsum = lsum * corr + w;
      mmax = mnew;
#pragma unroll
      for (int c = 0; c < 4; ++c) {
        v8bf vc = vsh[base + c];
#pragma unroll
        for (int e = 0; e < 8; ++e)
          ov[c * 8 + e] = ov[c * 8 + e] * corr + w * (float)vc[e];
      }
    }
  }

  float inv = 1.0f / lsum;
#pragma unroll
  for (int c = 0; c < 4; ++c) {
    v8bf oc;
#pragma unroll
    for (int e = 0; e < 8; ++e) oc[e] = (bf16)(ov[c * 8 + e] * inv);
    ((v8bf*)ob)[qi + c] = oc;
  }
}

// ---------------- host-side launcher ----------------------------------------
extern "C" void kernel_launch(void* const* d_in, const int* in_sizes, int n_in,
                              void* d_out, int out_size, void* d_ws, size_t ws_size,
                              hipStream_t stream) {
  const float* x   = (const float*)d_in[0];
  const float* y   = (const float*)d_in[1];
  const float* Wq  = (const float*)d_in[2];
  const float* bq  = (const float*)d_in[3];
  const float* Wk  = (const float*)d_in[4];
  const float* bk  = (const float*)d_in[5];
  const float* Wv  = (const float*)d_in[6];
  const float* bv  = (const float*)d_in[7];
  const float* rpb = (const float*)d_in[8];
  const float* Wp  = (const float*)d_in[9];
  const float* bp  = (const float*)d_in[10];
  float* out = (float*)d_out;

  // workspace layout (bytes)
  char* ws = (char*)d_ws;
  const size_t sz_act = (size_t)M_ * C_ * sizeof(bf16);   // 4 MiB
  const size_t sz_w   = (size_t)C_ * C_ * sizeof(bf16);   // 128 KiB
  bf16* xb  = (bf16*)(ws);
  bf16* yb  = (bf16*)(ws + sz_act);
  bf16* WqT = (bf16*)(ws + 2 * sz_act);
  bf16* WkT = (bf16*)(ws + 2 * sz_act + sz_w);
  bf16* WvT = (bf16*)(ws + 2 * sz_act + 2 * sz_w);
  bf16* WpT = (bf16*)(ws + 2 * sz_act + 3 * sz_w);
  bf16* qB  = (bf16*)(ws + 2 * sz_act + 4 * sz_w);
  bf16* kB  = (bf16*)(ws + 3 * sz_act + 4 * sz_w);
  bf16* vB  = (bf16*)(ws + 4 * sz_act + 4 * sz_w);
  bf16* aB  = (bf16*)(ws + 5 * sz_act + 4 * sz_w);

  const int n8 = M_ * C_ / 8;                 // 262144
  f32_to_bf16_vec<<<(n8 + 255) / 256, 256, 0, stream>>>(x, xb, n8);
  f32_to_bf16_vec<<<(n8 + 255) / 256, 256, 0, stream>>>(y, yb, n8);

  dim3 tb(16, 16), tg(16, 16);
  transpose_to_bf16_256<<<tg, tb, 0, stream>>>(Wq, WqT);
  transpose_to_bf16_256<<<tg, tb, 0, stream>>>(Wk, WkT);
  transpose_to_bf16_256<<<tg, tb, 0, stream>>>(Wv, WvT);
  transpose_to_bf16_256<<<tg, tb, 0, stream>>>(Wp, WpT);

  // waves = (M/32)*(N/64) = 256*4 = 1024 ; 8 waves/block -> 128 blocks
  const int gemm_blocks = (M_ / 32) * (C_ / 64) / 8;
  const float scale = 0.17677669529663687f;   // HD^-0.5
  gemm_wmma_bf16<true><<<gemm_blocks, 256, 0, stream>>>(xb, WqT, bq, qB, M_, C_, scale);
  gemm_wmma_bf16<true><<<gemm_blocks, 256, 0, stream>>>(yb, WkT, bk, kB, M_, C_, 1.0f);
  gemm_wmma_bf16<true><<<gemm_blocks, 256, 0, stream>>>(yb, WvT, bv, vB, M_, C_, 1.0f);

  // one block per (b, head, i): 64 query threads, K/V window staged in LDS
  nat_attention<<<B_ * NH_ * H_, 64, 0, stream>>>(qB, kB, vB, rpb, aB);

  // final projection, f32 output
  gemm_wmma_bf16<false><<<gemm_blocks, 256, 0, stream>>>(aB, WpT, bp, out, M_, C_, 1.0f);
}
